// MoEGate_38379827757773
// MI455X (gfx1250) — matrix operand order repaired and checked by reading, not compiled, not run on hardware
//
#include <hip/hip_runtime.h>

// ---------------- problem constants (from reference) ----------------
#define TOP_K       8
#define N_EXPERTS   256
#define N_GROUP     8
#define GROUP_SZ    32
#define TOPK_GROUP  4
#define HIDDEN      7168
#define TOKENS      16384
#define SCALE_F     2.5f

// ---------------- tiling ----------------
#define M_BLK   64                  // tokens per workgroup
#define KC      64                  // K chunk per iteration (2 WMMA K-steps)
#define NCHUNK  (HIDDEN / KC)       // 112
#define WS      36                  // LDS row stride in bf16-pairs (conflict-free vs 64 banks)
#define SS      257                 // score LDS row stride (floats)

typedef __attribute__((ext_vector_type(16))) __bf16    v16bf;
typedef __attribute__((ext_vector_type(8)))  float     v8f;
typedef __attribute__((ext_vector_type(4)))  float     f32x4;   // native vector: OK for nontemporal builtin
typedef __attribute__((ext_vector_type(4)))  unsigned  u32x4;

struct Frag32B { u32x4 lo, hi; };   // 32 B == one v16bf fragment

// Double-buffered GEMM staging, dead after the K loop -> union with epilogue scratch
struct StageBufs {
    unsigned A[2][M_BLK * WS];      // 2 x  9.2 KB
    unsigned W[2][N_EXPERTS * WS];  // 2 x 36.9 KB
};
struct EpiBufs {
    float grp[M_BLK][N_GROUP];      // group scores
    float cv [M_BLK][4 * TOP_K];    // per-quarter sorted top-8 values
    int   ci [M_BLK][4 * TOP_K];    // per-quarter sorted top-8 indices
};
union ScratchU { StageBufs g; EpiBufs e; };   // 92.2 KB

// fp32x2 -> packed bf16x2 via hardware convert (v_cvt_bf16_f32 / pk form)
__device__ __forceinline__ unsigned pk2(float a, float b) {
    union { __bf16 h[2]; unsigned u; } x;
    x.h[0] = (__bf16)a;
    x.h[1] = (__bf16)b;
    return x.u;
}

__global__ __launch_bounds__(256)
void moe_gate_kernel(const float* __restrict__ hidden,
                     const float* __restrict__ weight,
                     const float* __restrict__ bias,
                     float* __restrict__ out_idx,
                     float* __restrict__ out_w)
{
    __shared__ ScratchU sU;                 // 92.2 KB
    __shared__ float    sScore[M_BLK * SS]; // 65.8 KB
    __shared__ float    sBias[N_EXPERTS];   //  1.0 KB  (total ~155 KB -> 2 blocks/WGP)

    const int tid  = threadIdx.x;
    const int lane = tid & 31;              // wave32
    const int wave = tid >> 5;              // 8 waves
    const int half = lane >> 4;
    const int l16  = lane & 15;
    const int tok0 = blockIdx.x * M_BLK;
    const int mt   = wave >> 1;             // M-tile 0..3
    const int ebas = (wave & 1) * 128;      // expert half for this wave

    sBias[tid] = bias[tid];                 // blockDim == 256 == N_EXPERTS

    v8f acc[8];
    #pragma unroll
    for (int t = 0; t < 8; ++t)
        acc[t] = (v8f){0.f, 0.f, 0.f, 0.f, 0.f, 0.f, 0.f, 0.f};

    // software-pipeline registers: next chunk's raw fp32 data in flight across compute
    f32x4 wr[16];   // weight: 16 float4 per thread (256 rows x 64K / 256 thr)
    f32x4 ar[4];    // hidden:  4 float4 per thread (64 rows x 64K / 256 thr)

    auto load_regs = [&](int k0) {
        #pragma unroll
        for (int j = 0; j < 8; ++j) {              // one 8-float octet per step
            int i = tid + 256 * j;                 // 0..2047 octets
            int e = i >> 3, q = i & 7;
            const f32x4* p = (const f32x4*)&weight[(size_t)e * HIDDEN + k0 + q * 8];
            wr[2 * j]     = p[0];
            wr[2 * j + 1] = p[1];
        }
        #pragma unroll
        for (int j = 0; j < 2; ++j) {
            int i = tid + 256 * j;                 // 0..511 octets
            int r = i >> 3, q = i & 7;
            const f32x4* p = (const f32x4*)&hidden[(size_t)(tok0 + r) * HIDDEN + k0 + q * 8];
            ar[2 * j]     = __builtin_nontemporal_load(p);     // streamed once: keep L2 for weight
            ar[2 * j + 1] = __builtin_nontemporal_load(p + 1);
        }
    };

    auto pack_store = [&](int b) {
        #pragma unroll
        for (int j = 0; j < 8; ++j) {
            int i = tid + 256 * j;
            int e = i >> 3, q = i & 7;
            u32x4 u;
            u.x = pk2(wr[2 * j][0],     wr[2 * j][1]);
            u.y = pk2(wr[2 * j][2],     wr[2 * j][3]);
            u.z = pk2(wr[2 * j + 1][0], wr[2 * j + 1][1]);
            u.w = pk2(wr[2 * j + 1][2], wr[2 * j + 1][3]);
            *(u32x4*)&sU.g.W[b][e * WS + q * 4] = u;           // ds_store_b128
        }
        #pragma unroll
        for (int j = 0; j < 2; ++j) {
            int i = tid + 256 * j;
            int r = i >> 3, q = i & 7;
            u32x4 u;
            u.x = pk2(ar[2 * j][0],     ar[2 * j][1]);
            u.y = pk2(ar[2 * j][2],     ar[2 * j][3]);
            u.z = pk2(ar[2 * j + 1][0], ar[2 * j + 1][1]);
            u.w = pk2(ar[2 * j + 1][2], ar[2 * j + 1][3]);
            *(u32x4*)&sU.g.A[b][r * WS + q * 4] = u;
        }
    };

    auto compute = [&](int b) {
        #pragma unroll
        for (int s = 0; s < 2; ++s) {
            // A fragment (ISA A layout: K half-interleaved)
            const unsigned* ap = &sU.g.A[b][(mt * 16 + l16) * WS + s * 16 + half * 4];
            Frag32B fa;
            fa.lo = *(const u32x4*)(ap);
            fa.hi = *(const u32x4*)(ap + 8);
            v16bf A = __builtin_bit_cast(v16bf, fa);
            #pragma unroll
            for (int t = 0; t < 8; ++t) {
                // B fragment (same K half-interleaved layout as A)
                const unsigned* bp = &sU.g.W[b][(ebas + t * 16 + l16) * WS + s * 16 + half * 4];
                Frag32B fb;
                fb.lo = *(const u32x4*)(bp);
                fb.hi = *(const u32x4*)(bp + 8);
                v16bf B = __builtin_bit_cast(v16bf, fb);
                acc[t] = __builtin_amdgcn_wmma_f32_16x16x32_bf16(
                    false, A, false, B, (short)0, acc[t], false, false);
            }
        }
    };

    // ---- pipelined K loop: loads(c+1) -> wmma(c) -> pack/store(c+1) -> barrier ----
    load_regs(0);
    pack_store(0);
    __syncthreads();
    for (int c = 0; c < NCHUNK; ++c) {
        int cur = c & 1;
        if (c + 1 < NCHUNK) load_regs((c + 1) * KC);   // issue, no wait
        compute(cur);                                   // WMMA under load shadow
        if (c + 1 < NCHUNK) pack_store(cur ^ 1);        // wait loads, cvt, ds_store
        __syncthreads();
    }

    // ---- epilogue 1: sigmoid -> LDS ----
    #pragma unroll
    for (int t = 0; t < 8; ++t) {
        #pragma unroll
        for (int v = 0; v < 8; ++v) {
            int m = mt * 16 + v + 8 * half;
            int e = ebas + t * 16 + l16;
            sScore[m * SS + e] = 1.0f / (1.0f + __expf(-acc[t][v]));
        }
    }
    __syncthreads();

    // ---- epilogue 2: grouped top-k, 4 threads per token ----
    const int m = tid >> 2;          // token 0..63
    const int r = tid & 3;           // quarter: experts [r*64, r*64+64)
    const float* row = &sScore[m * SS];

    // group scores: sum of top-2 corrected scores (this thread owns groups 2r, 2r+1)
    #pragma unroll
    for (int gg = 0; gg < 2; ++gg) {
        int g = r * 2 + gg;
        float m1 = -1e30f, m2 = -1e30f;
        for (int j = 0; j < GROUP_SZ; ++j) {
            float v = row[g * GROUP_SZ + j] + sBias[g * GROUP_SZ + j];
            if (v > m1) { m2 = m1; m1 = v; }
            else if (v > m2) { m2 = v; }
        }
        sU.e.grp[m][g] = m1 + m2;
    }
    __syncthreads();

    // top-4 groups (computed redundantly by the token's 4 threads; first-index tie-break)
    unsigned gmask = 0;
    {
        float gs[N_GROUP];
        #pragma unroll
        for (int g = 0; g < N_GROUP; ++g) gs[g] = sU.e.grp[m][g];
        for (int i = 0; i < TOPK_GROUP; ++i) {
            float best = -1e38f; int bg = 0;
            #pragma unroll
            for (int g = 0; g < N_GROUP; ++g)
                if (!((gmask >> g) & 1u) && gs[g] > best) { best = gs[g]; bg = g; }
            gmask |= 1u << bg;
        }
    }

    // local sorted top-8 over this thread's 64 experts (stable: ties keep lower index)
    float tv[TOP_K]; int ti[TOP_K];
    #pragma unroll
    for (int q = 0; q < TOP_K; ++q) { tv[q] = -1e38f; ti[q] = 1 << 30; }
    for (int j = 0; j < 64; ++j) {
        int e = r * 64 + j;
        if (!((gmask >> (e >> 5)) & 1u)) continue;
        float v = row[e] + sBias[e];
        if (v > tv[TOP_K - 1]) {
            int pos = TOP_K - 1;
            while (pos > 0 && tv[pos - 1] < v) {
                tv[pos] = tv[pos - 1]; ti[pos] = ti[pos - 1]; --pos;
            }
            tv[pos] = v; ti[pos] = e;
        }
    }
    #pragma unroll
    for (int q = 0; q < TOP_K; ++q) {
        sU.e.cv[m][r * TOP_K + q] = tv[q];
        sU.e.ci[m][r * TOP_K + q] = ti[q];
    }
    __syncthreads();

    // exact 4-way merge of sorted lists -> global top-8 in descending order
    if (r == 0) {
        const float* cv = sU.e.cv[m];
        const int*   ci = sU.e.ci[m];
        int ptr[4] = {0, 8, 16, 24};
        int idx[TOP_K]; float wvv[TOP_K]; float wsum = 0.f;
        for (int i = 0; i < TOP_K; ++i) {
            float best = -1e38f; int bi = 1 << 30, bl = 0;
            #pragma unroll
            for (int l = 0; l < 4; ++l) {
                int pp = ptr[l];
                if (pp < (l + 1) * 8) {
                    float v = cv[pp]; int e2 = ci[pp];
                    if (v > best || (v == best && e2 < bi)) { best = v; bi = e2; bl = l; }
                }
            }
            ptr[bl]++;
            idx[i] = bi;
            float w = row[bi];          // uncorrected sigmoid score
            wvv[i] = w; wsum += w;
        }
        float norm = SCALE_F / (wsum + 1e-20f);
        int tg = tok0 + m;
        #pragma unroll
        for (int i = 0; i < TOP_K; ++i) {
            out_idx[(size_t)tg * TOP_K + i] = (float)idx[i];
            out_w  [(size_t)tg * TOP_K + i] = wvv[i] * norm;
        }
    }
}

extern "C" void kernel_launch(void* const* d_in, const int* in_sizes, int n_in,
                              void* d_out, int out_size, void* d_ws, size_t ws_size,
                              hipStream_t stream) {
    const float* hidden = (const float*)d_in[0];   // [4,4096,7168] f32
    const float* weight = (const float*)d_in[1];   // [256,7168]    f32
    const float* bias   = (const float*)d_in[2];   // [256]         f32

    float* out     = (float*)d_out;                // [T*8] idx (as float), then [T*8] weights
    float* out_idx = out;
    float* out_w   = out + (size_t)TOKENS * TOP_K;

    dim3 grid(TOKENS / M_BLK);   // 256 blocks
    dim3 block(256);             // 8 wave32 waves
    moe_gate_kernel<<<grid, block, 0, stream>>>(hidden, weight, bias, out_idx, out_w);
}
